// MultiLayerTemporalGraph_89584427860679
// MI455X (gfx1250) — matrix operand order repaired and checked by reading, not compiled, not run on hardware
//
#include <hip/hip_runtime.h>
#include <hip/hip_bf16.h>

// ---------------------------------------------------------------------------
// Problem constants (from reference)
// ---------------------------------------------------------------------------
#define BB   32
#define TT   60
#define NN   50
#define HH   512
#define LL   2
#define M0   (BB * TT * NN)          // 96000 rows (b,t,n)
#define M1   (BB * (TT - 1) * NN)    // 94400 rows (b,t<59,n)
#define AW_ELEMS  ((size_t)BB * TT * NN * NN)   // 4,800,000
#define COS_ELEMS ((size_t)M1)                  // 94,400

typedef __attribute__((ext_vector_type(16))) _Float16 v16h;
typedef __attribute__((ext_vector_type(8)))  float    v8f;

// ---------------------------------------------------------------------------
// Wave32 reductions
// ---------------------------------------------------------------------------
__device__ __forceinline__ float waveSum(float v) {
    #pragma unroll
    for (int off = 16; off > 0; off >>= 1) v += __shfl_xor(v, off, 32);
    return v;
}
__device__ __forceinline__ float waveMax(float v) {
    #pragma unroll
    for (int off = 16; off > 0; off >>= 1) v = fmaxf(v, __shfl_xor(v, off, 32));
    return v;
}

// ---------------------------------------------------------------------------
// WMMA GEMM:  C[M,O] = epilogue( A[M,K](f16) * Wh[O,K]^T(f16) )
// Block = 1 M-tile (16 rows) x 128 O-columns (8 waves, one 16x16 tile each).
// The 16xK f16 A stripe is staged into LDS with gfx1250 async-to-LDS loads
// (ASYNCcnt), then each wave streams B fragments from the L2-resident f16
// weights and issues v_wmma_f32_16x16x32_f16 with fp32 accumulation.
//   mode 0: out = acc + bias[o]
//   mode 1: out = relu(acc + bias[o])
//   mode 2: out = res[m,o] + relu(acc)                      (layer residual)
//   mode 3: out = res[curRow(m),o] + res2[m,o] + relu(acc + bias[o])   (upd)
// Requires: K multiple of 128, M multiple of 16, O == 128 * (gridDim.x mod),
// grid.x = (M/16) * (O/128) exactly (no remainder waves).
// ---------------------------------------------------------------------------
__global__ __launch_bounds__(256) void wmma_gemm_kernel(
    const _Float16* __restrict__ A, const _Float16* __restrict__ Wh,
    const float* __restrict__ bias, float* __restrict__ Cout,
    const float* __restrict__ res,  const float* __restrict__ res2,
    int M, int K, int O, int mode)
{
    __shared__ __align__(16) _Float16 As[16 * HH];   // 16 rows x K(<=512) halves

    const int lane  = threadIdx.x & 31;
    const int wid   = threadIdx.x >> 5;
    const int oGrp  = O >> 7;                        // O-groups of 128
    const int mBase = (blockIdx.x / oGrp) << 4;
    const int oBase = ((blockIdx.x % oGrp) << 7) + (wid << 4);
    const int hi = lane >> 4;                        // half-wave select
    const int lr = lane & 15;

    // ---- async-stage the 16xK A stripe into LDS (16*K*2 bytes) ----
    {
        const int t  = threadIdx.x;
        const int r  = t >> 4;                       // row 0..15
        const int cb = (t & 15) * (K >> 3);          // byte offset within row
        const char* gsrc = (const char*)(A + (size_t)(mBase + r) * K) + cb;
        const uint32_t ldsOff =
            (uint32_t)(uintptr_t)((char*)As + r * (K * 2) + cb);
        for (int q = 0; q < (K >> 7); ++q) {         // (K*2/16)/16 chunks of 16B
            const uint32_t la = ldsOff + q * 16;
            const uint64_t ga = (uint64_t)(uintptr_t)(gsrc + q * 16);
            asm volatile("global_load_async_to_lds_b128 %0, %1, off"
                         :: "v"(la), "v"(ga) : "memory");
        }
        asm volatile("s_wait_asynccnt 0x0" ::: "memory");
    }
    __syncthreads();

    v8f acc = {0.f, 0.f, 0.f, 0.f, 0.f, 0.f, 0.f, 0.f};

    const _Float16* __restrict__ arow = As + lr * K;                      // LDS
    const _Float16* __restrict__ wrow = Wh + (size_t)(oBase + lr) * K + hi * 16;

    for (int k0 = 0; k0 < K; k0 += 32) {
        __builtin_prefetch(wrow + k0 + 64, 0, 0);    // global_prefetch_b8
        // A fragment (LDS): element e -> K = k0 + hi*16 + e
        v16h a;
        {
            const _Float16* p0 = arow + k0 + hi * 16;
            #pragma unroll
            for (int e = 0; e < 8; ++e) { a[e] = p0[e]; a[e + 8] = p0[8 + e]; }
        }
        // B fragment (global f16 weights): B[k][n] = Wh[oBase+n][k0 + hi*16 + e]
        v16h bf;
        {
            const _Float16* q0 = wrow + k0;
            #pragma unroll
            for (int e = 0; e < 16; ++e) bf[e] = q0[e];
        }
        acc = __builtin_amdgcn_wmma_f32_16x16x32_f16(
                  false, a, false, bf, (short)0, acc, false, false);
    }

    const int o = oBase + lr;
    const float bv = bias ? bias[o] : 0.f;
    #pragma unroll
    for (int j = 0; j < 8; ++j) {
        const int m = mBase + j + hi * 8;            // C/D layout: VGPR j -> M=j(+8)
        const float v = acc[j];
        float outv;
        if (mode == 0)      outv = v + bv;
        else if (mode == 1) outv = fmaxf(v + bv, 0.f);
        else if (mode == 2) outv = res[(size_t)m * O + o] + fmaxf(v, 0.f);
        else {              // mode 3: upd = cur + tc + relu(ts_acc + bs)
            const int b = m / (NN * (TT - 1));
            const int curRow = m + NN * (b + 1);
            outv = res[(size_t)curRow * O + o] + res2[(size_t)m * O + o]
                 + fmaxf(v + bv, 0.f);
        }
        Cout[(size_t)m * O + o] = outv;
    }
}

// ---------------------------------------------------------------------------
// Pack concat(visual, audio) rows into f16 for the input projection GEMM
// ---------------------------------------------------------------------------
__global__ void pack_input_kernel(const float* __restrict__ visual,
                                  const float* __restrict__ audio,
                                  _Float16* __restrict__ Xh)
{
    const size_t total = (size_t)M0 * HH;
    for (size_t idx = (size_t)blockIdx.x * blockDim.x + threadIdx.x;
         idx < total; idx += (size_t)gridDim.x * blockDim.x) {
        const int d  = (int)(idx & (HH - 1));
        const int m  = (int)(idx >> 9);
        const int n  = m % NN;
        const int bt = m / NN;
        const float v = (n < NN - 1)
            ? visual[((size_t)bt * (NN - 1) + n) * HH + d]
            : audio[(size_t)bt * HH + d];
        Xh[idx] = (_Float16)v;
    }
}

__global__ void pack_f16_kernel(const float* __restrict__ src,
                                _Float16* __restrict__ dst, size_t total)
{
    for (size_t idx = (size_t)blockIdx.x * blockDim.x + threadIdx.x;
         idx < total; idx += (size_t)gridDim.x * blockDim.x)
        dst[idx] = (_Float16)src[idx];
}

// ---------------------------------------------------------------------------
// Per-(b,t) attention scores + softmax + threshold -> aw (written to d_out)
// x tile (50x512 f32) staged in dynamic LDS; row norms computed in-kernel.
// ---------------------------------------------------------------------------
__global__ __launch_bounds__(256) void scores_softmax_kernel(
    const float* __restrict__ x, float* __restrict__ awOut)
{
    extern __shared__ char smemRaw[];
    float* xs = (float*)smemRaw;              // 50*512 floats
    float* rn = xs + NN * HH;                 // 50 inverse norms (padded)
    const int tid  = threadIdx.x;
    const int lane = tid & 31;
    const int wid  = tid >> 5;
    const int blk  = blockIdx.x;              // (b*T + t)
    const size_t base = (size_t)blk * NN * HH;

    {
        const float4* src4 = (const float4*)(x + base);
        float4* xs4 = (float4*)xs;
        for (int i = tid; i < NN * (HH / 4); i += 256) xs4[i] = src4[i];
    }
    __syncthreads();

    for (int i = wid; i < NN; i += 8) {
        const float4* xi = (const float4*)(xs + i * HH) + lane * 4;
        float s = 0.f;
        #pragma unroll
        for (int c = 0; c < 4; ++c) {
            float4 a = xi[c];
            s += a.x * a.x + a.y * a.y + a.z * a.z + a.w * a.w;
        }
        s = waveSum(s);
        if (lane == 0) rn[i] = 1.f / fmaxf(sqrtf(s), 1e-12f);
    }
    __syncthreads();

    const float4* X4 = (const float4*)xs;
    for (int i = wid; i < NN; i += 8) {
        const int j1 = lane, j2 = lane + 32;
        const bool v2ok = (j2 < NN);
        float s1 = 0.f, s2 = 0.f;
        const float4* xi = X4 + i * (HH / 4);
        const float4* xa = X4 + j1 * (HH / 4);
        const float4* xb = X4 + (v2ok ? j2 : 0) * (HH / 4);
        for (int c = 0; c < HH / 4; ++c) {
            float4 a = xi[c];
            float4 b1 = xa[c];
            s1 += a.x * b1.x + a.y * b1.y + a.z * b1.z + a.w * b1.w;
            if (v2ok) {
                float4 b2 = xb[c];
                s2 += a.x * b2.x + a.y * b2.y + a.z * b2.z + a.w * b2.w;
            }
        }
        float sc1 = s1 * rn[i] * rn[j1];
        if (j1 == i) sc1 = -1e9f;
        float sc2 = v2ok ? s2 * rn[i] * rn[j2] : -1e30f;
        if (v2ok && j2 == i) sc2 = -1e9f;

        float mx = waveMax(fmaxf(sc1, sc2));
        const float e1 = expf(sc1 - mx);
        const float e2 = v2ok ? expf(sc2 - mx) : 0.f;
        const float inv = 1.f / waveSum(e1 + e2);

        float* awRow = awOut + (size_t)blk * NN * NN + (size_t)i * NN;
        {
            float a1 = e1 * inv;
            const float th = (i < NN - 1 && j1 < NN - 1) ? 0.021f : 0.032f;
            awRow[j1] = (a1 > th) ? a1 : 0.f;
        }
        if (v2ok) {
            float a2 = e2 * inv;
            const float th = (i < NN - 1 && j2 < NN - 1) ? 0.021f : 0.032f;
            awRow[j2] = (a2 > th) ? a2 : 0.f;
        }
    }
}

// ---------------------------------------------------------------------------
// sp = relu(layernorm(aw @ x + x)) per (b,t); wave-per-row, x & aw in LDS
// ---------------------------------------------------------------------------
__global__ __launch_bounds__(256) void sp_layernorm_kernel(
    const float* __restrict__ x, const float* __restrict__ aw,
    const float* __restrict__ g, const float* __restrict__ beta,
    float* __restrict__ spOut)
{
    extern __shared__ char smemRaw[];
    float* xs  = (float*)smemRaw;             // 50*512
    float* aws = xs + NN * HH;                // 50*50
    const int tid  = threadIdx.x;
    const int lane = tid & 31;
    const int wid  = tid >> 5;
    const int blk  = blockIdx.x;
    const size_t base = (size_t)blk * NN * HH;

    {
        const float4* src4 = (const float4*)(x + base);
        float4* xs4 = (float4*)xs;
        for (int i = tid; i < NN * (HH / 4); i += 256) xs4[i] = src4[i];
        const float* asrc = aw + (size_t)blk * NN * NN;
        for (int i = tid; i < NN * NN; i += 256) aws[i] = asrc[i];
    }
    __syncthreads();

    const int h0 = lane * 16;
    for (int i = wid; i < NN; i += 8) {
        float v[16];
        const float* xr = xs + i * HH + h0;
        #pragma unroll
        for (int e = 0; e < 16; ++e) v[e] = xr[e];
        for (int j = 0; j < NN; ++j) {
            const float c = aws[i * NN + j];    // wave-uniform broadcast
            if (c != 0.f) {
                const float* xj = xs + j * HH + h0;
                #pragma unroll
                for (int e = 0; e < 16; ++e) v[e] = fmaf(c, xj[e], v[e]);
            }
        }
        float s = 0.f, ss = 0.f;
        #pragma unroll
        for (int e = 0; e < 16; ++e) { s += v[e]; ss += v[e] * v[e]; }
        s = waveSum(s); ss = waveSum(ss);
        const float mu  = s * (1.f / HH);
        const float var = ss * (1.f / HH) - mu * mu;
        const float inv = rsqrtf(var + 1e-5f);
        float* dst = spOut + base + (size_t)i * HH + h0;
        #pragma unroll
        for (int e = 0; e < 16; ++e) {
            const float val = (v[e] - mu) * inv * g[h0 + e] + beta[h0 + e];
            dst[e] = fmaxf(val, 0.f);
        }
    }
}

// ---------------------------------------------------------------------------
// cosine(sp[:,1:], sp[:,:-1]) per node row; wave per row
// ---------------------------------------------------------------------------
__global__ __launch_bounds__(256) void cos_kernel(
    const float* __restrict__ sp, float* __restrict__ cosBuf,
    float* __restrict__ cosOut)
{
    const int lane = threadIdx.x & 31;
    const int wid  = threadIdx.x >> 5;
    const int m = blockIdx.x * 8 + wid;
    if (m >= M1) return;
    const int b = m / (NN * (TT - 1));
    const float* p = sp + (size_t)(m + NN * b) * HH + lane * 16;   // prev
    const float* a = p + (size_t)NN * HH;                          // cur
    float dot = 0.f, na = 0.f, nb = 0.f;
    #pragma unroll
    for (int e = 0; e < 16; ++e) {
        const float av = a[e], pv = p[e];
        dot += av * pv; na += av * av; nb += pv * pv;
    }
    dot = waveSum(dot); na = waveSum(na); nb = waveSum(nb);
    if (lane == 0) {
        const float c = dot / (fmaxf(sqrtf(na), 1e-8f) * fmaxf(sqrtf(nb), 1e-8f));
        cosBuf[m] = c;
        cosOut[m] = c;
    }
}

// ---------------------------------------------------------------------------
// A1 = f16(cos * prev), A2 = f16((1-cos) * cur)
// ---------------------------------------------------------------------------
__global__ void scale_pack_kernel(const float* __restrict__ sp,
                                  const float* __restrict__ cosv,
                                  _Float16* __restrict__ A1,
                                  _Float16* __restrict__ A2)
{
    const size_t total = (size_t)M1 * HH;
    for (size_t idx = (size_t)blockIdx.x * blockDim.x + threadIdx.x;
         idx < total; idx += (size_t)gridDim.x * blockDim.x) {
        const int h = (int)(idx & (HH - 1));
        const int m = (int)(idx >> 9);
        const int b = m / (NN * (TT - 1));
        const size_t prev = (size_t)(m + NN * b) * HH + h;
        const float c = cosv[m];
        A1[idx] = (_Float16)(c * sp[prev]);
        A2[idx] = (_Float16)((1.f - c) * sp[prev + (size_t)NN * HH]);
    }
}

// ---------------------------------------------------------------------------
// cat = concat(sp[:, :1], upd) ; layernorm(cat) -> f16 (GEMM A operand)
// ---------------------------------------------------------------------------
__global__ __launch_bounds__(256) void cat_ln_pack_kernel(
    const float* __restrict__ sp, const float* __restrict__ upd,
    const float* __restrict__ g, const float* __restrict__ beta,
    _Float16* __restrict__ out)
{
    const int lane = threadIdx.x & 31;
    const int wid  = threadIdx.x >> 5;
    const int r = blockIdx.x * 8 + wid;
    if (r >= M0) return;
    const int b = r / (TT * NN);
    const int t = (r / NN) % TT;
    const float* src = (t == 0)
        ? (sp  + (size_t)r * HH)
        : (upd + (size_t)(r - NN * b - NN) * HH);
    const int h0 = lane * 16;
    float v[16];
    float s = 0.f, ss = 0.f;
    #pragma unroll
    for (int e = 0; e < 16; ++e) {
        v[e] = src[h0 + e];
        s += v[e]; ss += v[e] * v[e];
    }
    s = waveSum(s); ss = waveSum(ss);
    const float mu  = s * (1.f / HH);
    const float var = ss * (1.f / HH) - mu * mu;
    const float inv = rsqrtf(var + 1e-5f);
    _Float16* dst = out + (size_t)r * HH + h0;
    #pragma unroll
    for (int e = 0; e < 16; ++e)
        dst[e] = (_Float16)((v[e] - mu) * inv * g[h0 + e] + beta[h0 + e]);
}

// ---------------------------------------------------------------------------
// Host launch
// ---------------------------------------------------------------------------
extern "C" void kernel_launch(void* const* d_in, const int* in_sizes, int n_in,
                              void* d_out, int out_size, void* d_ws, size_t ws_size,
                              hipStream_t stream)
{
    const float* visual = (const float*)d_in[0];
    const float* audio  = (const float*)d_in[1];
    const float* Win    = (const float*)d_in[2];
    const float* bin    = (const float*)d_in[3];
    const float* norm_g = (const float*)d_in[4];
    const float* norm_b = (const float*)d_in[5];
    const float* Wc     = (const float*)d_in[6];
    const float* bc     = (const float*)d_in[7];
    const float* Ws     = (const float*)d_in[8];
    const float* bs     = (const float*)d_in[9];
    const float* Wout   = (const float*)d_in[10];
    const float* Wfin   = (const float*)d_in[11];
    const float* bfin   = (const float*)d_in[12];

    float* out    = (float*)d_out;
    float* awOut  = out + (size_t)M0 * HH;
    float* cosOut = awOut + AW_ELEMS;

    // ---- workspace carve (256B aligned) ----
    char* w = (char*)d_ws;
    size_t off = 0;
    auto carve = [&](size_t bytes) -> void* {
        void* p = w + off;
        off = (off + bytes + 255) & ~(size_t)255;
        return p;
    };
    float*     nodeBuf = (float*)    carve((size_t)M0 * HH * sizeof(float));
    float*     spBuf   = (float*)    carve((size_t)M0 * HH * sizeof(float));
    float*     updBuf  = (float*)    carve((size_t)M1 * HH * sizeof(float));
    float*     cosBuf  = (float*)    carve((size_t)M1 * sizeof(float));
    _Float16*  hA      = (_Float16*) carve((size_t)M0 * HH * sizeof(_Float16));
    _Float16*  hB      = (_Float16*) carve((size_t)M1 * HH * sizeof(_Float16));
    _Float16*  Wh8     = (_Float16*) carve((size_t)8 * HH * HH * sizeof(_Float16));

    const size_t WSZ = (size_t)HH * HH;          // elements per weight matrix
    _Float16* WinH  = Wh8 + 0 * WSZ;
    _Float16* WfinH = Wh8 + 7 * WSZ;

    const int gemmBlocksM0 = (M0 / 16) * (HH / 128);   // 24000 (exact)
    const int gemmBlocksM1 = (M1 / 16) * (HH / 128);   // 23600 (exact)
    const size_t smScores = (size_t)NN * HH * sizeof(float) + 64 * sizeof(float);
    const size_t smSp     = (size_t)NN * HH * sizeof(float) + (size_t)NN * NN * sizeof(float);

    // ---- pre-pack all weight matrices to f16 (L2-resident afterwards) ----
    pack_f16_kernel<<<256, 256, 0, stream>>>(Win,  WinH,  WSZ);
    for (int l = 0; l < LL; ++l) {
        pack_f16_kernel<<<256, 256, 0, stream>>>(Wc   + l * WSZ, Wh8 + (1 + 3 * l) * WSZ, WSZ);
        pack_f16_kernel<<<256, 256, 0, stream>>>(Ws   + l * WSZ, Wh8 + (2 + 3 * l) * WSZ, WSZ);
        pack_f16_kernel<<<256, 256, 0, stream>>>(Wout + l * WSZ, Wh8 + (3 + 3 * l) * WSZ, WSZ);
    }
    pack_f16_kernel<<<256, 256, 0, stream>>>(Wfin, WfinH, WSZ);

    // input projection: node = concat(visual,audio) @ Win^T + bin
    pack_input_kernel<<<4096, 256, 0, stream>>>(visual, audio, hA);
    wmma_gemm_kernel<<<gemmBlocksM0, 256, 0, stream>>>(
        hA, WinH, bin, nodeBuf, nullptr, nullptr, M0, HH, HH, 0);

    for (int l = 0; l < LL; ++l) {
        const float* g_l  = norm_g + (size_t)l * HH;
        const float* b_l  = norm_b + (size_t)l * HH;
        const float* bc_l = bc + (size_t)l * HH;
        const float* bs_l = bs + (size_t)l * HH;
        _Float16* WcH = Wh8 + (1 + 3 * l) * WSZ;
        _Float16* WsH = Wh8 + (2 + 3 * l) * WSZ;
        _Float16* WoH = Wh8 + (3 + 3 * l) * WSZ;

        scores_softmax_kernel<<<BB * TT, 256, smScores, stream>>>(nodeBuf, awOut);
        sp_layernorm_kernel<<<BB * TT, 256, smSp, stream>>>(nodeBuf, awOut, g_l, b_l, spBuf);
        cos_kernel<<<(M1 + 7) / 8, 256, 0, stream>>>(spBuf, cosBuf, cosOut);
        scale_pack_kernel<<<4096, 256, 0, stream>>>(spBuf, cosBuf, hA, hB);
        // tc = relu((cos*prev) @ Wc^T + bc)
        wmma_gemm_kernel<<<gemmBlocksM1, 256, 0, stream>>>(
            hA, WcH, bc_l, updBuf, nullptr, nullptr, M1, HH, HH, 1);
        // upd = cur + tc + relu((1-cos)*cur @ Ws^T + bs)
        wmma_gemm_kernel<<<gemmBlocksM1, 256, 0, stream>>>(
            hB, WsH, bs_l, updBuf, spBuf, updBuf, M1, HH, HH, 3);
        // cat = LN(concat(sp[:, :1], upd)) -> f16
        cat_ln_pack_kernel<<<(M0 + 7) / 8, 256, 0, stream>>>(spBuf, updBuf, g_l, b_l, hA);
        // node = node + relu(cat @ Wout^T)
        wmma_gemm_kernel<<<gemmBlocksM0, 256, 0, stream>>>(
            hA, WoH, nullptr, nodeBuf, nodeBuf, nullptr, M0, HH, HH, 2);
    }

    // final: out = node @ Wfin^T + bfin
    pack_f16_kernel<<<4096, 256, 0, stream>>>(nodeBuf, hA, (size_t)M0 * HH);
    wmma_gemm_kernel<<<gemmBlocksM0, 256, 0, stream>>>(
        hA, WfinH, bfin, out, nullptr, nullptr, M0, HH, HH, 0);
}
